// FlattenInterCycleMoELayer_51616916963605
// MI455X (gfx1250) — matrix operand order repaired and checked by reading, not compiled, not run on hardware
//
#include <hip/hip_runtime.h>
#include <hip/hip_bf16.h>

#define B_SZ  2048
#define D_SZ  128
#define F_SZ  12800          // L*D = 100*128
#define DLLM_ 768
#define DFF_  256
#define NE    8
#define NOUT  (9 * D_SZ)     // 8 experts + general, column-concatenated

typedef __attribute__((ext_vector_type(16))) __bf16           v16bf;
typedef __attribute__((ext_vector_type(8)))  float            v8f;
typedef __attribute__((ext_vector_type(16))) unsigned short   v16us;
typedef __attribute__((ext_vector_type(8)))  unsigned short   v8us;

// Native f32 -> bf16 convert (gfx1250 has hardware bf16 cvt; avoid the
// 3-VALU-op software round sequence).
static __device__ __forceinline__ unsigned short f32_to_bf16(float f) {
    __bf16 b = (__bf16)f;
    return __builtin_bit_cast(unsigned short, b);
}
// Pack two converts; compiler can fuse into v_cvt_pk_bf16_f32.
static __device__ __forceinline__ unsigned int pack2_bf16(float a, float b) {
    return (unsigned int)f32_to_bf16(a) | ((unsigned int)f32_to_bf16(b) << 16);
}

static __device__ __forceinline__ float fast_tanh(float y) {
#if __has_builtin(__builtin_amdgcn_tanhf)
    return __builtin_amdgcn_tanhf(y);          // native V_TANH_F32 (trans op)
#else
    return 1.0f - 2.0f / (__expf(2.0f * y) + 1.0f);   // branch-free fallback
#endif
}

static __device__ __forceinline__ v16bf frag_from_2x8(const unsigned short* p_lo,
                                                      const unsigned short* p_hi) {
    v8us lo = *(const v8us*)p_lo;
    v8us hi = *(const v8us*)p_hi;
    v16us full;
#pragma unroll
    for (int i = 0; i < 8; ++i) { full[i] = lo[i]; full[8 + i] = hi[i]; }
    return __builtin_bit_cast(v16bf, full);
}

// ---------------------------------------------------------------------------
// Kernel 1: h[B,256] = gelu( concat(DKP, cycle, xf) @ Wg1 + bg1 )
// Block: 256 thr (8 waves); block tile 16 rows x 256 cols; wave: 2 N-tiles.
// K loop: 24 steps over DKP (768) + 400 steps over xf (12800); 'cycle'
// (gin row 768) applied as a rank-1 VALU update in the epilogue.
// ---------------------------------------------------------------------------
__global__ void __launch_bounds__(256)
gate_h_kernel(const float* __restrict__ xf, const float* __restrict__ cyc,
              const float* __restrict__ dkp, const float* __restrict__ Wg1,
              const float* __restrict__ bg1, float* __restrict__ h)
{
    __shared__ __align__(16) unsigned short lgA[16][40];   // 16x32 bf16, padded
    const int tid  = threadIdx.x;
    const int lane = tid & 31;
    const int wv   = tid >> 5;
    const int m0   = blockIdx.x * 16;
    const int nb   = wv * 32;

    v8f acc0 = {}; v8f acc1 = {};

    for (int s = 0; s < 24 + 400; ++s) {
        // ---- stage A chunk (16 rows x 32 K) as bf16 into LDS ----
        {
            int idx = tid * 2;                 // two consecutive elements
            int row = idx >> 5, kk = idx & 31;
            const float* src = (s < 24)
                ? &dkp[(size_t)(m0 + row) * DLLM_ + (s * 32 + kk)]
                : &xf [(size_t)(m0 + row) * F_SZ  + ((s - 24) * 32 + kk)];
            *(unsigned int*)&lgA[row][kk] = pack2_bf16(src[0], src[1]);
        }
        __syncthreads();

        const int wrow = (s < 24) ? (s * 32) : (DLLM_ + 1 + (s - 24) * 32);

        // A fragment (contiguous K): lanes<16 -> K 0..15; else K 16..31
        const int Mloc = lane & 15;
        const int akb  = (lane < 16) ? 0 : 16;
        v16bf afrag = frag_from_2x8(&lgA[Mloc][akb], &lgA[Mloc][akb + 8]);

        // B fragments straight from global Wg1 (no cross-wave reuse in block)
        const int bkb = (lane < 16) ? 0 : 16;
#pragma unroll
        for (int t = 0; t < 2; ++t) {
            int N = nb + t * 16 + (lane & 15);
            v16us bfull;
#pragma unroll
            for (int hh = 0; hh < 16; ++hh)
                bfull[hh] = f32_to_bf16(Wg1[(size_t)(wrow + bkb + hh) * DFF_ + N]);
            v16bf bfrag = __builtin_bit_cast(v16bf, bfull);
            if (t == 0)
                acc0 = __builtin_amdgcn_wmma_f32_16x16x32_bf16(false, afrag, false, bfrag,
                                                               (short)0, acc0, false, false);
            else
                acc1 = __builtin_amdgcn_wmma_f32_16x16x32_bf16(false, afrag, false, bfrag,
                                                               (short)0, acc1, false, false);
        }
        __syncthreads();
    }

    // epilogue: rank-1 cycle term + bias + gelu(tanh); C layout: vgpr r -> M=r(+8)
#pragma unroll
    for (int t = 0; t < 2; ++t) {
        v8f acc = (t == 0) ? acc0 : acc1;
        int N = nb + t * 16 + (lane & 15);
        float wcyc = Wg1[(size_t)DLLM_ * DFF_ + N];    // gin row 768
        float bias = bg1[N];
#pragma unroll
        for (int r = 0; r < 8; ++r) {
            int M = r + ((lane < 16) ? 0 : 8);
            float x = acc[r] + cyc[m0 + M] * wcyc + bias;
            float g = 0.5f * x * (1.0f + fast_tanh(0.7978845608028654f *
                                                   (x + 0.044715f * x * x * x)));
            h[(size_t)(m0 + M) * DFF_ + N] = g;
        }
    }
}

// ---------------------------------------------------------------------------
// Kernel 2: logits -> top-2 masked-softmax gates [B,8]   (tiny; VALU)
// ---------------------------------------------------------------------------
__global__ void __launch_bounds__(256)
gate_route_kernel(const float* __restrict__ h, const float* __restrict__ Wg2,
                  const float* __restrict__ bg2, float* __restrict__ gates)
{
    int b = blockIdx.x * blockDim.x + threadIdx.x;
    if (b >= B_SZ) return;
    float logit[NE];
#pragma unroll
    for (int e = 0; e < NE; ++e) logit[e] = bg2[e];
    for (int j = 0; j < DFF_; ++j) {
        float hv = h[(size_t)b * DFF_ + j];
#pragma unroll
        for (int e = 0; e < NE; ++e) logit[e] += hv * Wg2[j * NE + e];
    }
    int i1 = 0;
#pragma unroll
    for (int e = 1; e < NE; ++e) if (logit[e] > logit[i1]) i1 = e;
    int i2 = (i1 == 0) ? 1 : 0;
#pragma unroll
    for (int e = 0; e < NE; ++e) if (e != i1 && logit[e] > logit[i2]) i2 = e;
    float mx = logit[i1], p[NE], s = 0.f;
#pragma unroll
    for (int e = 0; e < NE; ++e) { p[e] = __expf(logit[e] - mx); s += p[e]; }
    float denom = (p[i1] + p[i2]) / s + 1e-9f;
#pragma unroll
    for (int e = 0; e < NE; ++e) {
        float g = (e == i1 || e == i2) ? ((p[e] / s) / denom) : 0.f;
        gates[b * NE + e] = g;
    }
}

// ---------------------------------------------------------------------------
// Kernel 3: Y[B, 9*128] = xf @ [We[0]|...|We[7]|Wgen]     (60.4 GFLOP)
// Grid (B/128, 9); block 256 thr = 8 waves (2 M-halves x 4 N-quarters);
// wave tile 64x32 = 4x2 WMMA tiles. K staged 64 per round (2 WMMA K-steps,
// 16 WMMAs between barriers). A/B fit L2 (178 MB < 192 MB) so tile re-reads
// stay on-chip; HBM traffic ~= compulsory 185 MB.
// ---------------------------------------------------------------------------
__global__ void __launch_bounds__(256)
expert_gemm_kernel(const float* __restrict__ xf, const float* __restrict__ We,
                   const float* __restrict__ Wgen, float* __restrict__ Y)
{
    __shared__ __align__(16) unsigned short lgA[128][72];  // [m][k] bf16, 64K + pad
    __shared__ __align__(16) unsigned short lgB[128][72];  // [n][k] bf16 (transposed)
    const int tid  = threadIdx.x;
    const int lane = tid & 31;
    const int wv   = tid >> 5;
    const int wm   = wv & 1;          // M half   (0..1) -> 64 rows
    const int wn   = wv >> 1;         // N quarter(0..3) -> 32 cols
    const int m0   = blockIdx.x * 128;
    const int nb   = blockIdx.y;      // expert id; 8 == general
    const float* Wsrc = (nb < NE) ? (We + (size_t)nb * F_SZ * D_SZ) : Wgen;

    v8f acc[4][2];
#pragma unroll
    for (int i = 0; i < 4; ++i)
#pragma unroll
        for (int j = 0; j < 2; ++j) acc[i][j] = (v8f){};

    const int NITER = F_SZ / 64;      // 200
    for (int s = 0; s < NITER; ++s) {
        const int k0 = s * 64;
        // ---- stage A (128 x 64): float4 loads, packed bf16 pair stores ----
#pragma unroll
        for (int i = 0; i < 8; ++i) {
            int v   = i * 256 + tid;          // float4 index, 2048 total
            int row = v >> 4;                 // 16 float4 per row
            int kk  = (v & 15) * 4;
            float4 f = *(const float4*)&xf[(size_t)(m0 + row) * F_SZ + k0 + kk];
            uint2 u; u.x = pack2_bf16(f.x, f.y); u.y = pack2_bf16(f.z, f.w);
            *(uint2*)&lgA[row][kk] = u;
        }
        // ---- stage B (64 x 128) transposed into [n][k] ----
#pragma unroll
        for (int i = 0; i < 8; ++i) {
            int v  = i * 256 + tid;
            int kr = v >> 5;                  // 32 float4 per k-row
            int n  = (v & 31) * 4;
            float4 f = *(const float4*)&Wsrc[(size_t)(k0 + kr) * D_SZ + n];
            lgB[n + 0][kr] = f32_to_bf16(f.x);
            lgB[n + 1][kr] = f32_to_bf16(f.y);
            lgB[n + 2][kr] = f32_to_bf16(f.z);
            lgB[n + 3][kr] = f32_to_bf16(f.w);
        }
        // prefetch next K tile (emits global_prefetch_b8; warms WGP$/L2)
        if (s + 1 < NITER) {
            __builtin_prefetch(&xf[(size_t)(m0 + (tid >> 1)) * F_SZ + k0 + 64 +
                                   (tid & 1) * 32], 0, 0);
            __builtin_prefetch(&Wsrc[(size_t)(k0 + 64 + (tid >> 2)) * D_SZ +
                                     (tid & 3) * 32], 0, 0);
        }
        __syncthreads();

        const int akb = (lane < 16) ? 0 : 16;
        const int bkb = (lane < 16) ? 0 : 16;
#pragma unroll
        for (int kq = 0; kq < 64; kq += 32) {
            v16bf bfrag[2];
#pragma unroll
            for (int tn = 0; tn < 2; ++tn) {
                int N = wn * 32 + tn * 16 + (lane & 15);
                bfrag[tn] = frag_from_2x8(&lgB[N][kq + bkb], &lgB[N][kq + bkb + 8]);
            }
#pragma unroll
            for (int tm = 0; tm < 4; ++tm) {
                int M = wm * 64 + tm * 16 + (lane & 15);
                v16bf afrag = frag_from_2x8(&lgA[M][kq + akb], &lgA[M][kq + akb + 8]);
#pragma unroll
                for (int tn = 0; tn < 2; ++tn)
                    acc[tm][tn] = __builtin_amdgcn_wmma_f32_16x16x32_bf16(
                        false, afrag, false, bfrag[tn], (short)0, acc[tm][tn], false, false);
            }
        }
        __syncthreads();
    }

    // epilogue: C layout vgpr r -> row r (+8 for lanes>=16), col = lane%16
#pragma unroll
    for (int tm = 0; tm < 4; ++tm)
#pragma unroll
        for (int tn = 0; tn < 2; ++tn) {
            int Ncol = nb * D_SZ + wn * 32 + tn * 16 + (lane & 15);
#pragma unroll
            for (int r = 0; r < 8; ++r) {
                int Mrow = m0 + wm * 64 + tm * 16 + r + ((lane < 16) ? 0 : 8);
                Y[(size_t)Mrow * NOUT + Ncol] = acc[tm][tn][r];
            }
        }
}

// ---------------------------------------------------------------------------
// Kernel 4: out = (general + bgen) + bf16round( sum_e gates_e*(Ye + be_e) )
// ---------------------------------------------------------------------------
__global__ void __launch_bounds__(256)
combine_kernel(const float* __restrict__ Y, const float* __restrict__ gates,
               const float* __restrict__ be, const float* __restrict__ bgen,
               float* __restrict__ out)
{
    int t = blockIdx.x * blockDim.x + threadIdx.x;
    if (t >= B_SZ * D_SZ) return;
    int b = t >> 7, d = t & 127;
    const float* yr = Y + (size_t)b * NOUT;
    float comb = 0.f;
#pragma unroll
    for (int e = 0; e < NE; ++e)
        comb += gates[b * NE + e] * (yr[e * D_SZ + d] + be[e * D_SZ + d]);
    float combf = __bfloat162float(__float2bfloat16(comb));   // reference bf16 cast
    out[t] = (yr[NE * D_SZ + d] + bgen[d]) + combf;
}

extern "C" void kernel_launch(void* const* d_in, const int* in_sizes, int n_in,
                              void* d_out, int out_size, void* d_ws, size_t ws_size,
                              hipStream_t stream) {
    const float* xf   = (const float*)d_in[0];   // cycle_curve_data [B,100,128] == [B,F]
    const float* cyc  = (const float*)d_in[1];   // [B,1]
    const float* dkp  = (const float*)d_in[2];   // [B,768]
    const float* Wg1  = (const float*)d_in[3];   // [13569,256]
    const float* bg1  = (const float*)d_in[4];   // [256]
    const float* Wg2  = (const float*)d_in[5];   // [256,8]
    const float* bg2  = (const float*)d_in[6];   // [8]
    const float* We   = (const float*)d_in[7];   // [8,12800,128]
    const float* be   = (const float*)d_in[8];   // [8,128]
    const float* Wgen = (const float*)d_in[9];   // [12800,128]
    const float* bgen = (const float*)d_in[10];  // [128]
    float* out = (float*)d_out;

    float* h     = (float*)d_ws;                        // B*256 f32
    float* gates = h + (size_t)B_SZ * DFF_;             // B*8   f32
    float* Y     = gates + (size_t)B_SZ * NE;           // B*1152 f32  (~11.6 MB total)

    gate_h_kernel     <<<B_SZ / 16,           256, 0, stream>>>(xf, cyc, dkp, Wg1, bg1, h);
    gate_route_kernel <<<B_SZ / 256,          256, 0, stream>>>(h, Wg2, bg2, gates);
    expert_gemm_kernel<<<dim3(B_SZ / 128, 9), 256, 0, stream>>>(xf, We, Wgen, Y);
    combine_kernel    <<<(B_SZ * D_SZ) / 256, 256, 0, stream>>>(Y, gates, be, bgen, out);
}
